// RGCN_7533372637979
// MI455X (gfx1250) — matrix operand order repaired and checked by reading, not compiled, not run on hardware
//
#include <hip/hip_runtime.h>
#include <hip/hip_bf16.h>

// ---------------------------------------------------------------------------
// RGCN on MI455X (gfx1250). Strategy: transform-then-aggregate.
//   h0 = relu(feat @ win + bin)                     (WMMA bf16 GEMM)
//   per layer: hB = h @ loop + hbias                (WMMA GEMM, fp32 out)
//              for r: Yr = h @ W_r (bf16)           (WMMA GEMM)
//                     edges(etype==r): hB[dst] += Yr[src]  (f32 atomics, L2)
//              h' = relu(hB) -> bf16
//   out = relu(h @ wout + bout)                     (WMMA GEMM, 64 cols)
// ---------------------------------------------------------------------------

typedef __attribute__((ext_vector_type(16))) __bf16 v16bf;
typedef __attribute__((ext_vector_type(8)))  float  v8f;

#define HDIM 128
#define RREL 8
#define BBAS 4

static __device__ __forceinline__ unsigned short f2bf(float f) {
    unsigned u = __float_as_uint(f);
    unsigned r = u + 0x7FFFu + ((u >> 16) & 1u);   // round-to-nearest-even
    return (unsigned short)(r >> 16);
}
static __device__ __forceinline__ float bf2f(unsigned short h) {
    return __uint_as_float(((unsigned)h) << 16);
}

// --------------------------- prep kernels ----------------------------------

__global__ __launch_bounds__(256) void k_f32_to_bf16(const float* __restrict__ x,
                                                     unsigned short* __restrict__ y,
                                                     int n4) {
    int i = blockIdx.x * blockDim.x + threadIdx.x;
    if (i >= n4) return;
    float4 v = ((const float4*)x)[i];
    uint2 p;
    p.x = (unsigned)f2bf(v.x) | ((unsigned)f2bf(v.y) << 16);
    p.y = (unsigned)f2bf(v.z) | ((unsigned)f2bf(v.w) << 16);
    ((uint2*)y)[i] = p;
}

__global__ __launch_bounds__(256) void k_relu_to_bf16(const float* __restrict__ x,
                                                      unsigned short* __restrict__ y,
                                                      int n4) {
    int i = blockIdx.x * blockDim.x + threadIdx.x;
    if (i >= n4) return;
    float4 v = ((const float4*)x)[i];
    v.x = fmaxf(v.x, 0.f); v.y = fmaxf(v.y, 0.f);
    v.z = fmaxf(v.z, 0.f); v.w = fmaxf(v.w, 0.f);
    uint2 p;
    p.x = (unsigned)f2bf(v.x) | ((unsigned)f2bf(v.y) << 16);
    p.y = (unsigned)f2bf(v.z) | ((unsigned)f2bf(v.w) << 16);
    ((uint2*)y)[i] = p;
}

// W [K][NC] fp32 -> WT [NC][K] bf16
__global__ __launch_bounds__(256) void k_transpose_bf16(const float* __restrict__ W,
                                                        unsigned short* __restrict__ WT,
                                                        int K, int NC) {
    int i = blockIdx.x * blockDim.x + threadIdx.x;
    if (i >= K * NC) return;
    int k = i / NC, n = i % NC;
    WT[n * K + k] = f2bf(W[k * NC + n]);
}

// W_r[i][o] = sum_b wcomp[r,b]*basis[b,i,o]; store transposed: WT[r][o][i]
__global__ __launch_bounds__(256) void k_combine_basis(const float* __restrict__ wcomp,
                                                       const float* __restrict__ basis,
                                                       unsigned short* __restrict__ WT) {
    int i = blockIdx.x * blockDim.x + threadIdx.x;
    if (i >= RREL * HDIM * HDIM) return;
    int r = i / (HDIM * HDIM);
    int rem = i % (HDIM * HDIM);
    int o = rem / HDIM, in = rem % HDIM;
    float v = 0.f;
#pragma unroll
    for (int b = 0; b < BBAS; ++b)
        v += wcomp[r * BBAS + b] * basis[((size_t)b * HDIM + in) * HDIM + o];
    WT[((size_t)r * HDIM + o) * HDIM + in] = f2bf(v);
}

// --------------------------- WMMA GEMM -------------------------------------
// C[M][NCOLS] = A[M][128](bf16) x W[128][NCOLS], W given transposed WT[NCOLS][128].
// 256 threads = 8 waves; wave w owns rows [blk*128 + w*16, +16), all NCOLS cols.
// Requires M % 16 == 0 for full-tile loads (N=100000 satisfies this).

template <int NCOLS>
__global__ __launch_bounds__(256) void k_gemm_wmma(
    const unsigned short* __restrict__ A,
    const unsigned short* __restrict__ WT,
    const float* __restrict__ bias,        // [NCOLS] or nullptr
    float* __restrict__ Cf,                // fp32 out or nullptr
    unsigned short* __restrict__ Cb,       // bf16 out or nullptr
    int M, int relu)
{
    constexpr int K = 128;
    constexpr int PADK = K + 8;            // +16B pad per row -> conflict-free-ish
    constexpr int NT = NCOLS / 16;
    __shared__ unsigned short lds_wt[NCOLS * PADK];

    // stage transposed weight into LDS (row n: K bf16, stride PADK)
    for (int c = threadIdx.x; c < NCOLS * (K / 8); c += 256) {
        int n = c >> 4;
        int kk = (c & 15) * 8;
        uint4 v = *(const uint4*)(WT + (size_t)n * K + kk);
        *(uint4*)(&lds_wt[n * PADK + kk]) = v;
    }
    __syncthreads();

    const int wave = threadIdx.x >> 5;
    const int lane = threadIdx.x & 31;
    const int row0 = blockIdx.x * 128 + wave * 16;
    if (row0 >= M) return;                 // wave-uniform: EXEC stays all-ones

    const int l15   = lane & 15;
    const int khalf = lane >> 4;           // 0 or 1
    const size_t arow = (size_t)(row0 + l15) * K;

    v8f acc[NT];
#pragma unroll
    for (int t = 0; t < NT; ++t)
#pragma unroll
        for (int j = 0; j < 8; ++j) acc[t][j] = 0.f;

#pragma unroll
    for (int k0 = 0; k0 < K; k0 += 32) {
        // A fragment 16x32 bf16: lane half holds K {k0+8h..+7, k0+16+8h..+7}
        union { v16bf v; uint4 u[2]; } af;
        const unsigned short* ap = A + arow + k0 + khalf * 8;
        af.u[0] = *(const uint4*)(ap);
        af.u[1] = *(const uint4*)(ap + 16);
#pragma unroll
        for (int t = 0; t < NT; ++t) {
            // B fragment 32x16: lane -> col n, K {k0+8h..+7, k0+16+8h..+7}
            union { v16bf v; uint4 u[2]; } bf;
            const unsigned short* bp =
                &lds_wt[(t * 16 + l15) * PADK + k0 + khalf * 8];
            bf.u[0] = *(const uint4*)(bp);
            bf.u[1] = *(const uint4*)(bp + 16);
            acc[t] = __builtin_amdgcn_wmma_f32_16x16x32_bf16(
                false, af.v, false, bf.v, (short)0, acc[t], false, false);
        }
    }

    // C/D layout: lanes 0-15 -> M=j, lanes 16-31 -> M=8+j; N = lane&15
    const int rbase = row0 + khalf * 8;
#pragma unroll
    for (int t = 0; t < NT; ++t) {
        const int n = t * 16 + l15;
        const float bv = bias ? bias[n] : 0.f;
#pragma unroll
        for (int j = 0; j < 8; ++j) {
            float v = acc[t][j] + bv;
            if (relu) v = fmaxf(v, 0.f);
            const size_t off = (size_t)(rbase + j) * NCOLS + n;
            if (Cf) Cf[off] = v;
            if (Cb) Cb[off] = f2bf(v);
        }
    }
}

// --------------------------- edge scatter ----------------------------------
// For edges with etype==r: acc[dst][:] += Yr[src][:]. One wave per edge,
// 4 bf16 gathered per lane, fp32 atomic adds into L2-resident accumulator.

__global__ __launch_bounds__(256) void k_edge_scatter(
    const int* __restrict__ src, const int* __restrict__ dst,
    const int* __restrict__ et, int r,
    const unsigned short* __restrict__ Y, float* __restrict__ acc, int E)
{
    const int lane = threadIdx.x & 31;
    const int wid  = (blockIdx.x * blockDim.x + threadIdx.x) >> 5;
    const int nw   = (gridDim.x * blockDim.x) >> 5;
    for (int e = wid; e < E; e += nw) {
        if (et[e] != r) continue;
        const int s = src[e];
        const int d = dst[e];
        const uint2 yv = *(const uint2*)(Y + (size_t)s * HDIM + lane * 4);
        float* ap = acc + (size_t)d * HDIM + lane * 4;
        __hip_atomic_fetch_add(ap + 0, bf2f((unsigned short)(yv.x & 0xFFFFu)),
                               __ATOMIC_RELAXED, __HIP_MEMORY_SCOPE_AGENT);
        __hip_atomic_fetch_add(ap + 1, bf2f((unsigned short)(yv.x >> 16)),
                               __ATOMIC_RELAXED, __HIP_MEMORY_SCOPE_AGENT);
        __hip_atomic_fetch_add(ap + 2, bf2f((unsigned short)(yv.y & 0xFFFFu)),
                               __ATOMIC_RELAXED, __HIP_MEMORY_SCOPE_AGENT);
        __hip_atomic_fetch_add(ap + 3, bf2f((unsigned short)(yv.y >> 16)),
                               __ATOMIC_RELAXED, __HIP_MEMORY_SCOPE_AGENT);
    }
}

// --------------------------- host orchestration ----------------------------

extern "C" void kernel_launch(void* const* d_in, const int* in_sizes, int n_in,
                              void* d_out, int out_size, void* d_ws, size_t ws_size,
                              hipStream_t stream) {
    const float* feat   = (const float*)d_in[0];
    const float* win    = (const float*)d_in[1];
    const float* bin_   = (const float*)d_in[2];
    const float* basis0 = (const float*)d_in[3];
    const float* wcomp0 = (const float*)d_in[4];
    const float* loop0  = (const float*)d_in[5];
    const float* hbias0 = (const float*)d_in[6];
    const float* basis1 = (const float*)d_in[7];
    const float* wcomp1 = (const float*)d_in[8];
    const float* loop1  = (const float*)d_in[9];
    const float* hbias1 = (const float*)d_in[10];
    const float* wout   = (const float*)d_in[11];
    const float* bout   = (const float*)d_in[12];
    const int*   srcI   = (const int*)d_in[13];
    const int*   dstI   = (const int*)d_in[14];
    const int*   etyI   = (const int*)d_in[15];

    const int N = in_sizes[0] / HDIM;     // 100000 (multiple of 16)
    const int E = in_sizes[13];           // 1600000
    const size_t NH = (size_t)N * HDIM;

    // workspace carve-out (256B aligned), total ~129 MB
    char* wp = (char*)d_ws;
    auto carve = [&](size_t bytes) {
        void* p = (void*)wp;
        wp += (bytes + 255) & ~(size_t)255;
        return p;
    };
    unsigned short* hb_even = (unsigned short*)carve(NH * 2); // featb -> h1b
    unsigned short* hb_odd  = (unsigned short*)carve(NH * 2); // h0b   -> h2b
    float*          hBf     = (float*)carve(NH * 4);          // fp32 accumulator
    unsigned short* Yr      = (unsigned short*)carve(NH * 2); // per-relation Y
    unsigned short* winT    = (unsigned short*)carve(HDIM * HDIM * 2);
    unsigned short* loop0T  = (unsigned short*)carve(HDIM * HDIM * 2);
    unsigned short* loop1T  = (unsigned short*)carve(HDIM * HDIM * 2);
    unsigned short* woutT   = (unsigned short*)carve(64 * HDIM * 2);
    unsigned short* W0T     = (unsigned short*)carve((size_t)RREL * HDIM * HDIM * 2);
    unsigned short* W1T     = (unsigned short*)carve((size_t)RREL * HDIM * HDIM * 2);
    (void)ws_size; (void)n_in; (void)out_size;

    const int n4 = (int)(NH / 4);
    const int eltBlocks = (n4 + 255) / 256;
    const int gemmBlocks = (N + 127) / 128;

    // prep: conversions + weight composition (transposed bf16)
    k_f32_to_bf16<<<eltBlocks, 256, 0, stream>>>(feat, hb_even, n4);
    k_transpose_bf16<<<(HDIM * HDIM + 255) / 256, 256, 0, stream>>>(win, winT, HDIM, HDIM);
    k_transpose_bf16<<<(HDIM * HDIM + 255) / 256, 256, 0, stream>>>(loop0, loop0T, HDIM, HDIM);
    k_transpose_bf16<<<(HDIM * HDIM + 255) / 256, 256, 0, stream>>>(loop1, loop1T, HDIM, HDIM);
    k_transpose_bf16<<<(HDIM * 64 + 255) / 256, 256, 0, stream>>>(wout, woutT, HDIM, 64);
    k_combine_basis<<<(RREL * HDIM * HDIM + 255) / 256, 256, 0, stream>>>(wcomp0, basis0, W0T);
    k_combine_basis<<<(RREL * HDIM * HDIM + 255) / 256, 256, 0, stream>>>(wcomp1, basis1, W1T);

    // input layer: h0 = relu(feat @ win + bin)  (bf16 out only)
    k_gemm_wmma<HDIM><<<gemmBlocks, 256, 0, stream>>>(
        hb_even, winT, bin_, nullptr, hb_odd, N, 1);

    // two RelGraphConv layers
    const unsigned short* cur = hb_odd;
    unsigned short*       nxt = hb_even;
    const unsigned short* WT[2]     = {W0T, W1T};
    const unsigned short* loopT[2]  = {loop0T, loop1T};
    const float*          hbias[2]  = {hbias0, hbias1};

    for (int l = 0; l < 2; ++l) {
        // self-loop + bias seeds the fp32 accumulator
        k_gemm_wmma<HDIM><<<gemmBlocks, 256, 0, stream>>>(
            cur, loopT[l], hbias[l], hBf, nullptr, N, 0);
        for (int r = 0; r < RREL; ++r) {
            k_gemm_wmma<HDIM><<<gemmBlocks, 256, 0, stream>>>(
                cur, WT[l] + (size_t)r * HDIM * HDIM, nullptr, nullptr, Yr, N, 0);
            k_edge_scatter<<<2048, 256, 0, stream>>>(srcI, dstI, etyI, r, Yr, hBf, E);
        }
        k_relu_to_bf16<<<eltBlocks, 256, 0, stream>>>(hBf, nxt, n4);
        const unsigned short* t = cur; cur = nxt; nxt = (unsigned short*)t;
    }

    // output layer: out = relu(h @ wout + bout), fp32 to d_out [N,64]
    k_gemm_wmma<64><<<gemmBlocks, 256, 0, stream>>>(
        cur, woutT, bout, (float*)d_out, nullptr, N, 1);
}
